// Net_53498112639136
// MI455X (gfx1250) — matrix operand order repaired and checked by reading, not compiled, not run on hardware
//
#include <hip/hip_runtime.h>

#define NN 16384
#define NE 131072
#define NE2 (NN + NE)
#define NG 64
#define CDIV(a,b) (((a)+(b)-1)/(b))

typedef __attribute__((ext_vector_type(16))) __bf16 v16bf;
typedef __attribute__((ext_vector_type(8)))  __bf16 v8bf;
typedef __attribute__((ext_vector_type(2)))  __bf16 v2bf;
typedef __attribute__((ext_vector_type(8)))  float  v8f;

// ---------------- helpers ----------------
__device__ __forceinline__ unsigned fenc(float f) {
    unsigned u = __float_as_uint(f);
    return (u & 0x80000000u) ? ~u : (u | 0x80000000u);
}
__device__ __forceinline__ float fdec(unsigned u) {
    unsigned b = (u & 0x80000000u) ? (u & 0x7FFFFFFFu) : ~u;
    return __uint_as_float(b);
}

__global__ void k_fill(float* __restrict__ p, float v, int n) {
    int i = blockIdx.x * blockDim.x + threadIdx.x;
    if (i < n) p[i] = v;
}
__global__ void k_fillu(unsigned* __restrict__ p, unsigned v, int n) {
    int i = blockIdx.x * blockDim.x + threadIdx.x;
    if (i < n) p[i] = v;
}
__global__ void k_axpy(float* __restrict__ out, const float* __restrict__ a, int n) {
    int i = blockIdx.x * blockDim.x + threadIdx.x;
    if (i < n) out[i] += a[i];
}
// pad columns: src[rows,F] -> dst[rows,Fpad] (zero-fill tail cols)
__global__ void k_pad_cols(const float* __restrict__ src, float* __restrict__ dst,
                           int F, int Fpad, int rows) {
    int i = blockIdx.x * blockDim.x + threadIdx.x;
    if (i >= rows * Fpad) return;
    int r = i / Fpad, c = i % Fpad;
    dst[i] = (c < F) ? src[(size_t)r * F + c] : 0.0f;
}
// pad rows: src[R,C] -> dst[Rpad,C] (zero-fill tail rows)
__global__ void k_pad_rows(const float* __restrict__ src, float* __restrict__ dst,
                           int R, int Rpad, int C) {
    int i = blockIdx.x * blockDim.x + threadIdx.x;
    if (i >= Rpad * C) return;
    int r = i / C;
    dst[i] = (r < R) ? src[i] : 0.0f;
}

// ---------------- WMMA bf16 GEMM: C[M,N] = A[M,K] @ W[K,N] (+bias, +relu) ----------------
// Requires K % 32 == 0, M % 128 == 0, N % 64 == 0 (caller pre-pads ragged K).
// 256 threads = 8 wave32. Block tile 128x64; wave w owns rows [16w,16w+16), all 64 cols:
// 1 A-fragment, 4 B-fragments, 4 accumulators -> 4 v_wmma_f32_16x16x32_bf16 per K-chunk.
// All fragments are preloaded (contiguous 16B runs -> ds_load_b128) so the 4 WMMAs issue
// back-to-back after a single s_wait_dscnt, overlapping XDL with next-chunk staging.
__global__ __launch_bounds__(256) void k_gemm_bf16(
    const float* __restrict__ A, const float* __restrict__ Wm,
    const float* __restrict__ bias, float* __restrict__ C,
    int K, int N, int applyRelu)
{
    __shared__ __bf16 As[128][40];   // row stride 80B: 16B aligned + bank spread
    __shared__ __bf16 BsT[64][40];   // transposed: BsT[n][k]

    const int tid  = threadIdx.x;
    const int wave = tid >> 5;
    const int lane = tid & 31;
    const int hl   = lane >> 4;      // lane half (K-split per WMMA layout)
    const int l16  = lane & 15;

    const int bm = blockIdx.y * 128;
    const int bn = blockIdx.x * 64;

    v8f acc[4] = {{}, {}, {}, {}};

    for (int k0 = 0; k0 < K; k0 += 32) {
        // stage A 128x32: thread -> 8 consecutive k of one row (two b128 global loads)
        for (int t = tid; t < 512; t += 256) {
            int r = t >> 2, kq = (t & 3) * 8;
            const float4* ap = (const float4*)(A + (size_t)(bm + r) * K + k0 + kq);
            float4 a0 = ap[0];
            float4 a1 = ap[1];
            v8bf pk;
            pk[0] = (__bf16)a0.x; pk[1] = (__bf16)a0.y; pk[2] = (__bf16)a0.z; pk[3] = (__bf16)a0.w;
            pk[4] = (__bf16)a1.x; pk[5] = (__bf16)a1.y; pk[6] = (__bf16)a1.z; pk[7] = (__bf16)a1.w;
            *(v8bf*)&As[r][kq] = pk;
        }
        // stage B 32x64 transposed: thread -> (2 consecutive k, 1 n); coalesced along n
        for (int t = tid; t < 1024; t += 256) {
            int kp = t >> 6, n = t & 63;
            const float* wp = Wm + (size_t)(k0 + 2 * kp) * N + bn + n;
            v2bf pk;
            pk[0] = (__bf16)wp[0];
            pk[1] = (__bf16)wp[N];
            *(v2bf*)&BsT[n][2 * kp] = pk;
        }
        __syncthreads();

        // A fragment: lane (hl,l16) -> row wave*16+l16, K runs [hl*8,+8) and [16+hl*8,+8)
        const int arow = wave * 16 + l16;
        v8bf alo = *(const v8bf*)&As[arow][hl * 8];
        v8bf ahi = *(const v8bf*)&As[arow][16 + hl * 8];
        v16bf af = __builtin_shufflevector(alo, ahi, 0,1,2,3,4,5,6,7,8,9,10,11,12,13,14,15);

        // preload ALL B fragments, then issue the 4 WMMAs back-to-back
        v16bf bf[4];
#pragma unroll
        for (int t = 0; t < 4; ++t) {
            v8bf blo = *(const v8bf*)&BsT[t * 16 + l16][hl * 8];
            v8bf bhi = *(const v8bf*)&BsT[t * 16 + l16][16 + hl * 8];
            bf[t] = __builtin_shufflevector(blo, bhi, 0,1,2,3,4,5,6,7,8,9,10,11,12,13,14,15);
        }
#pragma unroll
        for (int t = 0; t < 4; ++t) {
            acc[t] = __builtin_amdgcn_wmma_f32_16x16x32_bf16(false, af, false, bf[t],
                                                             (short)0, acc[t], false, false);
        }
        __syncthreads();
    }

    // C/D layout: VGPR j -> M = j + 8*half, N = lane%16
#pragma unroll
    for (int t = 0; t < 4; ++t) {
        int gn = bn + t * 16 + l16;
        float bv = bias ? bias[gn] : 0.0f;
#pragma unroll
        for (int j = 0; j < 8; ++j) {
            int gm = bm + wave * 16 + hl * 8 + j;
            float v = acc[t][j] + bv;
            if (applyRelu) v = fmaxf(v, 0.0f);
            C[(size_t)gm * N + gn] = v;
        }
    }
}

// ---------------- graph kernels ----------------
__global__ void k_deg(const int* __restrict__ ed, float* __restrict__ deg) {
    int i = blockIdx.x * blockDim.x + threadIdx.x;
    if (i >= NE2) return;
    int d = (i < NE) ? ed[i] : (i - NE);
    atomicAdd(&deg[d], 1.0f);
}
__global__ void k_dinv(const float* __restrict__ deg, float* __restrict__ dinv) {
    int i = blockIdx.x * blockDim.x + threadIdx.x;
    if (i >= NN) return;
    float v = deg[i];
    dinv[i] = (v > 0.0f) ? rsqrtf(v) : 0.0f;
}

__global__ void k_gcn_agg(const float* __restrict__ h, const int* __restrict__ es,
                          const int* __restrict__ ed, const float* __restrict__ dinv,
                          float* __restrict__ out, int F) {
    int e = blockIdx.x;
    int src, dst;
    if (e < NE) { src = es[e]; dst = ed[e]; } else { src = dst = e - NE; }
    float norm = dinv[src] * dinv[dst];
    const float* hp = h + (size_t)src * F;
    float* op = out + (size_t)dst * F;
    for (int f = threadIdx.x; f < F; f += blockDim.x)
        atomicAdd(&op[f], hp[f] * norm);
}

__global__ void k_gat_al(const float* __restrict__ h, const float* __restrict__ as,
                         const float* __restrict__ ad, float* __restrict__ al, int Cc) {
    int i = blockIdx.x * blockDim.x + threadIdx.x;
    if (i >= NN * 3) return;
    int node = i / 3, hd = i % 3;
    const float* hp  = h + (size_t)node * 3 * Cc + hd * Cc;
    const float* asp = as + hd * Cc;
    const float* adp = ad + hd * Cc;
    float ss = 0.0f, sd = 0.0f;
    for (int c = 0; c < Cc; ++c) { float v = hp[c]; ss = fmaf(v, asp[c], ss); sd = fmaf(v, adp[c], sd); }
    al[2 * i] = ss; al[2 * i + 1] = sd;
}
__global__ void k_gat_max(const float* __restrict__ al, const int* __restrict__ es,
                          const int* __restrict__ ed, unsigned* __restrict__ emaxU) {
    int i = blockIdx.x * blockDim.x + threadIdx.x;
    if (i >= NE2 * 3) return;
    int e = i / 3, hd = i % 3, src, dst;
    if (e < NE) { src = es[e]; dst = ed[e]; } else { src = dst = e - NE; }
    float v = al[2 * (src * 3 + hd)] + al[2 * (dst * 3 + hd) + 1];
    v = (v > 0.0f) ? v : 0.2f * v;                   // leaky_relu 0.2
    atomicMax(&emaxU[dst * 3 + hd], fenc(v));
}
__global__ void k_gat_exp(const float* __restrict__ al, const int* __restrict__ es,
                          const int* __restrict__ ed, const unsigned* __restrict__ emaxU,
                          float* __restrict__ ee, float* __restrict__ denom) {
    int i = blockIdx.x * blockDim.x + threadIdx.x;
    if (i >= NE2 * 3) return;
    int e = i / 3, hd = i % 3, src, dst;
    if (e < NE) { src = es[e]; dst = ed[e]; } else { src = dst = e - NE; }
    float v = al[2 * (src * 3 + hd)] + al[2 * (dst * 3 + hd) + 1];
    v = (v > 0.0f) ? v : 0.2f * v;
    float ev = __expf(v - fdec(emaxU[dst * 3 + hd]));
    ee[i] = ev;
    atomicAdd(&denom[dst * 3 + hd], ev);
}
__global__ void k_gat_agg(const float* __restrict__ h, const int* __restrict__ es,
                          const int* __restrict__ ed, const float* __restrict__ ee,
                          const float* __restrict__ denom, float* __restrict__ out, int Cc) {
    int e = blockIdx.x;
    int src, dst;
    if (e < NE) { src = es[e]; dst = ed[e]; } else { src = dst = e - NE; }
    int F = 3 * Cc;
    const float* hp = h + (size_t)src * F;
    float* op = out + (size_t)dst * F;
    for (int f = threadIdx.x; f < F; f += blockDim.x) {
        int hd = f / Cc;
        float alpha = ee[e * 3 + hd] / denom[dst * 3 + hd];
        atomicAdd(&op[f], hp[f] * alpha);
    }
}

__global__ void k_gin_agg(const float* __restrict__ xin, const int* __restrict__ es,
                          const int* __restrict__ ed, float* __restrict__ out, int F) {
    int e = blockIdx.x;
    int src = es[e], dst = ed[e];
    const float* hp = xin + (size_t)src * F;
    float* op = out + (size_t)dst * F;
    for (int f = threadIdx.x; f < F; f += blockDim.x)
        atomicAdd(&op[f], hp[f]);
}

// ---------------- pointwise / BN / pool / head ----------------
__global__ void k_bias_relu(float* __restrict__ p, const float* __restrict__ b, int n, int F) {
    int i = blockIdx.x * blockDim.x + threadIdx.x;
    if (i >= n) return;
    float v = p[i] + (b ? b[i % F] : 0.0f);
    p[i] = fmaxf(v, 0.0f);
}
__global__ void k_bn_stats(const float* __restrict__ xp, float* __restrict__ mean,
                           float* __restrict__ var, int F) {
    __shared__ float s1[256], s2[256];
    int f = blockIdx.x, tid = threadIdx.x;
    float a = 0.0f, b = 0.0f;
    for (int r = tid; r < NN; r += 256) {
        float v = xp[(size_t)r * F + f];
        a += v; b += v * v;
    }
    s1[tid] = a; s2[tid] = b;
    __syncthreads();
    for (int st = 128; st > 0; st >>= 1) {
        if (tid < st) { s1[tid] += s1[tid + st]; s2[tid] += s2[tid + st]; }
        __syncthreads();
    }
    if (tid == 0) {
        float mu = s1[0] * (1.0f / NN);
        mean[f] = mu;
        var[f]  = s2[0] * (1.0f / NN) - mu * mu;
    }
}
__global__ void k_bn_apply(float* __restrict__ p, const float* __restrict__ mean,
                           const float* __restrict__ var, const float* __restrict__ g,
                           const float* __restrict__ b, int n, int F) {
    int i = blockIdx.x * blockDim.x + threadIdx.x;
    if (i >= n) return;
    int f = i % F;
    p[i] = g[f] * (p[i] - mean[f]) * rsqrtf(var[f] + 1e-5f) + b[f];
}
__global__ void k_pool(const float* __restrict__ xp, const int* __restrict__ batch,
                       float* __restrict__ cr, int F, int colOff) {
    int i = blockIdx.x * blockDim.x + threadIdx.x;
    if (i >= NN * F) return;
    int r = i / F, c = i % F;
    atomicAdd(&cr[batch[r] * 576 + colOff + c], xp[i]);
}
__global__ void k_fc(const float* __restrict__ xp, const float* __restrict__ Wm,
                     const float* __restrict__ b, float* __restrict__ y,
                     int M, int K, int Nout) {
    int i = blockIdx.x * blockDim.x + threadIdx.x;
    if (i >= M * Nout) return;
    int r = i / Nout, c = i % Nout;
    float s = b[c];
    for (int k = 0; k < K; ++k) s = fmaf(xp[r * K + k], Wm[k * Nout + c], s);
    y[i] = fmaxf(s, 0.0f);
}

// ---------------- host orchestration ----------------
extern "C" void kernel_launch(void* const* d_in, const int* in_sizes, int n_in,
                              void* d_out, int out_size, void* d_ws, size_t ws_size,
                              hipStream_t stream) {
    (void)in_sizes; (void)n_in; (void)out_size; (void)ws_size;

    const float* x = (const float*)d_in[0];
    const float* gcn_W[3]  = {(const float*)d_in[1],  (const float*)d_in[3],  (const float*)d_in[5]};
    const float* gcn_b[3]  = {(const float*)d_in[2],  (const float*)d_in[4],  (const float*)d_in[6]};
    const float* gat_W[3]  = {(const float*)d_in[7],  (const float*)d_in[11], (const float*)d_in[15]};
    const float* gat_as[3] = {(const float*)d_in[8],  (const float*)d_in[12], (const float*)d_in[16]};
    const float* gat_ad[3] = {(const float*)d_in[9],  (const float*)d_in[13], (const float*)d_in[17]};
    const float* gat_b[3]  = {(const float*)d_in[10], (const float*)d_in[14], (const float*)d_in[18]};
    const float* gin_Wa[3] = {(const float*)d_in[19], (const float*)d_in[23], (const float*)d_in[27]};
    const float* gin_ba[3] = {(const float*)d_in[20], (const float*)d_in[24], (const float*)d_in[28]};
    const float* gin_Wb[3] = {(const float*)d_in[21], (const float*)d_in[25], (const float*)d_in[29]};
    const float* gin_bb[3] = {(const float*)d_in[22], (const float*)d_in[26], (const float*)d_in[30]};
    const float* bn_g[9]; const float* bn_b[9];
    for (int i = 0; i < 9; ++i) { bn_g[i] = (const float*)d_in[31 + 2 * i]; bn_b[i] = (const float*)d_in[32 + 2 * i]; }
    const float* fc1W = (const float*)d_in[49]; const float* fc1b = (const float*)d_in[50];
    const float* fc2W = (const float*)d_in[51]; const float* fc2b = (const float*)d_in[52];
    const float* fc3W = (const float*)d_in[53]; const float* fc3b = (const float*)d_in[54];
    const int* edges = (const int*)d_in[55];
    const int* batch = (const int*)d_in[56];
    const int* es = edges;
    const int* ed = edges + NE;

    float* w = (float*)d_ws;
    float* buf0 = w; w += (size_t)NN * 768;
    float* buf1 = w; w += (size_t)NN * 768;
    float* buf2 = w; w += (size_t)NN * 768;
    float* deg  = w; w += NN;
    float* dinv = w; w += NN;
    float* al   = w; w += NN * 6;
    float* ee   = w; w += (size_t)NE2 * 3;
    unsigned* emaxU = (unsigned*)w; w += NN * 3;
    float* denom = w; w += NN * 3;
    float* mean = w; w += 768;
    float* var  = w; w += 768;
    float* cr   = w; w += NG * 576;
    float* f1   = w; w += NG * 256;
    float* f2   = w; w += NG * 64;
    float* xp    = w; w += (size_t)NN * 384;   // x padded 373 -> 384 cols
    float* w1gcn = w; w += (size_t)384 * 256;  // K-padded layer-1 weights
    float* w1gat = w; w += (size_t)384 * 768;
    float* w1gin = w; w += (size_t)384 * 256;

    auto fill = [&](float* p, float v, size_t n) {
        k_fill<<<(unsigned)CDIV(n, 256), 256, 0, stream>>>(p, v, (int)n);
    };
    auto gemm = [&](const float* A, const float* Wm, const float* bias, float* C,
                    int K, int Nout, int relu) {
        dim3 g((unsigned)(Nout / 64), (unsigned)(NN / 128));
        k_gemm_bf16<<<g, 256, 0, stream>>>(A, Wm, bias, C, K, Nout, relu);
    };
    auto bn = [&](float* p, int F, int idx) {
        k_bn_stats<<<(unsigned)F, 256, 0, stream>>>(p, mean, var, F);
        k_bn_apply<<<(unsigned)CDIV((size_t)NN * F, 256), 256, 0, stream>>>(p, mean, var, bn_g[idx], bn_b[idx], NN * F, F);
    };
    auto brelu = [&](float* p, const float* b, int F) {
        k_bias_relu<<<(unsigned)CDIV((size_t)NN * F, 256), 256, 0, stream>>>(p, b, NN * F, F);
    };

    // one-time padding of ragged-K operands (K=373 -> 384)
    k_pad_cols<<<CDIV((size_t)NN * 384, 256), 256, 0, stream>>>(x, xp, 373, 384, NN);
    k_pad_rows<<<CDIV(384 * 256, 256), 256, 0, stream>>>(gcn_W[0], w1gcn, 373, 384, 256);
    k_pad_rows<<<CDIV(384 * 768, 256), 256, 0, stream>>>(gat_W[0], w1gat, 373, 384, 768);
    k_pad_rows<<<CDIV(384 * 256, 256), 256, 0, stream>>>(gin_Wa[0], w1gin, 373, 384, 256);

    // degrees (with self-loops) + concat buffer init
    fill(deg, 0.0f, NN);
    k_deg<<<CDIV(NE2, 256), 256, 0, stream>>>(ed, deg);
    k_dinv<<<CDIV(NN, 256), 256, 0, stream>>>(deg, dinv);
    fill(cr, 0.0f, (size_t)NG * 576);

    // -------- GCN tower --------
    auto gcn_layer = [&](const float* inp, const float* Wm, float* q, float* outb,
                         int K, int F, int li, int bni) {
        gemm(inp, Wm, nullptr, q, K, F, 0);
        fill(outb, 0.0f, (size_t)NN * F);
        k_gcn_agg<<<NE2, 128, 0, stream>>>(q, es, ed, dinv, outb, F);
        brelu(outb, gcn_b[li], F);
        bn(outb, F, bni);
    };
    gcn_layer(xp,   w1gcn,    buf0, buf1, 384, 256, 0, 0);
    gcn_layer(buf1, gcn_W[1], buf2, buf0, 256, 128, 1, 1);
    gcn_layer(buf0, gcn_W[2], buf2, buf1, 128, 128, 2, 2);
    k_pool<<<CDIV((size_t)NN * 128, 256), 256, 0, stream>>>(buf1, batch, cr, 128, 0);

    // -------- GAT tower --------
    auto gat_layer = [&](const float* inp, const float* Wm, float* q, float* outb,
                         int K, int Cc, int li, int bni) {
        int F = 3 * Cc;
        gemm(inp, Wm, nullptr, q, K, F, 0);
        k_gat_al<<<CDIV(NN * 3, 256), 256, 0, stream>>>(q, gat_as[li], gat_ad[li], al, Cc);
        k_fillu<<<CDIV(NN * 3, 256), 256, 0, stream>>>(emaxU, 0x007FFFFFu, NN * 3); // fenc(-inf)
        fill(denom, 0.0f, NN * 3);
        k_gat_max<<<CDIV(NE2 * 3, 256), 256, 0, stream>>>(al, es, ed, emaxU);
        k_gat_exp<<<CDIV(NE2 * 3, 256), 256, 0, stream>>>(al, es, ed, emaxU, ee, denom);
        fill(outb, 0.0f, (size_t)NN * F);
        k_gat_agg<<<NE2, 256, 0, stream>>>(q, es, ed, ee, denom, outb, Cc);
        brelu(outb, gat_b[li], F);
        bn(outb, F, bni);
    };
    gat_layer(xp,   w1gat,    buf0, buf1, 384, 256, 0, 3);
    gat_layer(buf1, gat_W[1], buf2, buf0, 768, 128, 1, 4);
    gat_layer(buf0, gat_W[2], buf2, buf1, 384, 128, 2, 5);
    k_pool<<<CDIV((size_t)NN * 384, 256), 256, 0, stream>>>(buf1, batch, cr, 384, 128);

    // -------- GIN tower --------
    auto gin_layer = [&](const float* inp, const float* Wa, int K, int Fh, int li, int bni) {
        fill(buf0, 0.0f, (size_t)NN * K);
        k_gin_agg<<<NE, 128, 0, stream>>>(inp, es, ed, buf0, K);
        k_axpy<<<CDIV((size_t)NN * K, 256), 256, 0, stream>>>(buf0, inp, NN * K);
        gemm(buf0, Wa, gin_ba[li], buf1, K, Fh, 1);           // relu(h@W1+b1)
        gemm(buf1, gin_Wb[li], gin_bb[li], buf2, Fh, Fh, 0);  // h@W2+b2
        brelu(buf2, nullptr, Fh);                             // outer relu
        bn(buf2, Fh, bni);
    };
    gin_layer(xp,   w1gin,     384, 256, 0, 6);   // padded cols are zeros: agg/axpy no-op there
    gin_layer(buf2, gin_Wa[1], 256, 128, 1, 7);
    gin_layer(buf2, gin_Wa[2], 128,  64, 2, 8);
    k_pool<<<CDIV((size_t)NN * 64, 256), 256, 0, stream>>>(buf2, batch, cr, 64, 512);

    // -------- head --------
    k_fc<<<CDIV(NG * 256, 256), 256, 0, stream>>>(cr, fc1W, fc1b, f1, NG, 576, 256);
    k_fc<<<CDIV(NG * 64, 64), 64, 0, stream>>>(f1, fc2W, fc2b, f2, NG, 256, 64);
    k_fc<<<1, 64, 0, stream>>>(f2, fc3W, fc3b, (float*)d_out, NG, 64, 1);
}
